// InstructBertEncoder_360777253109
// MI455X (gfx1250) — matrix operand order repaired and checked by reading, not compiled, not run on hardware
//
#include <hip/hip_runtime.h>

// ---------------------------------------------------------------------------
// InstructBERT attention for MI455X (gfx1250, wave32, WMMA).
// Pipeline: convert(fp32->bf16, W transpose) -> 5x projection GEMMs (WMMA
// bf16 16x16x32, f32 acc, async-LDS double-buffered weight tiles) -> fused
// attention kernel (scores + relative position bias via windowed dist_emb
// matmuls + softmax + PV + gated instruct branch). Workspace: ~43 MB.
// ---------------------------------------------------------------------------

typedef unsigned short u16;
typedef unsigned int   u32;
typedef __attribute__((ext_vector_type(16))) __bf16 v16bf;
typedef __attribute__((ext_vector_type(8)))  float  v8f;
typedef __attribute__((ext_vector_type(4)))  u32    u32x4;

union FragU { u32x4 q[2]; v16bf v; };

__device__ __forceinline__ u16 f2bf(float f) {
  union { float f; u32 u; } x; x.f = f;
  u32 r = x.u + 0x7fffu + ((x.u >> 16) & 1u);   // round-to-nearest-even
  return (u16)(r >> 16);
}

// A-fragment 16x32 bf16 (M x K), source row-major, row stride `ld` elements.
// lane L: row = L&15; lanes 0-15 hold K {0..7, 16..23}, lanes 16-31 hold
// K {8..15, 24..31} (per CDNA5 ISA 7.12.2).
__device__ __forceinline__ v16bf load_a_bf16(const u16* base, int ld) {
  const int lane = threadIdx.x & 31;
  const u16* p = base + (lane & 15) * ld + ((lane < 16) ? 0 : 8);
  FragU f;
  f.q[0] = *(const u32x4*)(p);
  f.q[1] = *(const u32x4*)(p + 16);
  return f.v;
}

// B-fragment 32x16 bf16 (K x N), source = B^T row-major (rows = N, contiguous
// K), row stride `ld`. lane L: col = L&15; lanes 0-15 hold K {0..7, 16..23},
// lanes 16-31 hold K {8..15, 24..31}.
__device__ __forceinline__ v16bf load_b_bf16(const u16* base, int ld) {
  const int lane = threadIdx.x & 31;
  const u16* p = base + (lane & 15) * ld + ((lane < 16) ? 0 : 8);
  FragU f;
  f.q[0] = *(const u32x4*)(p);
  f.q[1] = *(const u32x4*)(p + 16);
  return f.v;
}

__device__ __forceinline__ v8f wmma_bf(v16bf a, v16bf b, v8f c) {
  return __builtin_amdgcn_wmma_f32_16x16x32_bf16(false, a, false, b,
                                                 (short)0, c, false, false);
}

// Async copy: 16 bytes global -> LDS per lane (GLOBAL_LOAD_ASYNC_TO_LDS_B128,
// tracked by ASYNCcnt). LDS address = low 32 bits of generic LDS pointer
// (ISA 10.2).
__device__ __forceinline__ void async_copy_b128(const u16* lds_dst, const u16* g_src) {
  const u32 loff = (u32)(uintptr_t)(const void*)lds_dst;
  asm volatile("global_load_async_to_lds_b128 %0, %1, off"
               :: "v"(loff), "v"(g_src) : "memory");
}

__device__ __forceinline__ void wait_async0() {
  asm volatile("s_wait_asynccnt 0x0" ::: "memory");
}

// ---------------------------------------------------------------------------
// Kernel 1: fp32 -> bf16 conversions + weight transpose (Wt[n][k] = W[k][n]).
// ---------------------------------------------------------------------------
__global__ void convert_kernel(const float* __restrict__ X, const float* __restrict__ Xi,
                               const float* __restrict__ E,
                               const float* __restrict__ Wq, const float* __restrict__ Wk,
                               const float* __restrict__ Wv,
                               u16* __restrict__ Xb, u16* __restrict__ Xib,
                               u16* __restrict__ Eb, u16* __restrict__ Wtq,
                               u16* __restrict__ Wtk, u16* __restrict__ Wtv) {
  const int NX = 8 * 512 * 1024, NXI = 512 * 1024, NE = 1023 * 64, NW = 1024 * 1024;
  const int total = NX + NXI + NE + 3 * NW;
  for (int i = blockIdx.x * blockDim.x + threadIdx.x; i < total;
       i += gridDim.x * blockDim.x) {
    int idx = i;
    if (idx < NX)  { Xb[idx]  = f2bf(X[idx]);  continue; }
    idx -= NX;
    if (idx < NXI) { Xib[idx] = f2bf(Xi[idx]); continue; }
    idx -= NXI;
    if (idx < NE)  { Eb[idx]  = f2bf(E[idx]);  continue; }
    idx -= NE;
    const float* Wsrc = (idx < NW) ? Wq : ((idx < 2 * NW) ? Wk : Wv);
    u16*         Wdst = (idx < NW) ? Wtq : ((idx < 2 * NW) ? Wtk : Wtv);
    idx = idx % NW;
    const int nn = idx >> 10, kk = idx & 1023;
    Wdst[idx] = f2bf(Wsrc[kk * 1024 + nn]);
  }
}

// ---------------------------------------------------------------------------
// Kernel 2: projection GEMM. Block = 128(M) x 64(N) tile, 8 waves each own a
// 16-row strip. Weight k-slab (64x32 bf16, 4KB) is staged into LDS with
// async-to-LDS copies, double-buffered so the copy of slab k+1 overlaps the
// WMMAs on slab k. Stores bf16 head-major [B][H][L][64] (Q/K/IK) or
// transposed [B][H][64][L] (V/IV).
// ---------------------------------------------------------------------------
__global__ __launch_bounds__(256) void proj_kernel(
    const u16* __restrict__ Xb, const u16* __restrict__ Wt,
    const float* __restrict__ bias, u16* __restrict__ dst,
    int Lseg, int transposed) {
  __shared__ __align__(16) u16 sW[2][64 * 32];   // double-buffered Wt k-slab

  const int w = threadIdx.x >> 5, lane = threadIdx.x & 31;
  const int nt = blockIdx.x & 15, mtb = blockIdx.x >> 4;
  const int m0 = mtb * 128 + w * 16, n0g = nt * 64;

  // staging indices: 256 lanes x 16B cover the 64x32 slab
  const int srow = threadIdx.x >> 2, scol = (threadIdx.x & 3) * 8;

  auto stage = [&](int ks, int buf) {
    async_copy_b128(&sW[buf][srow * 32 + scol],
                    Wt + (size_t)(n0g + srow) * 1024 + ks * 32 + scol);
  };

  stage(0, 0);
  v8f acc[4] = {};
  for (int ks = 0; ks < 32; ks++) {
    wait_async0();        // own slab-ks portion arrived
    __syncthreads();      // all portions visible; prev-iter LDS reads retired
    if (ks < 31) stage(ks + 1, (ks + 1) & 1);
    const u16* Wl = &sW[ks & 1][0];
    v16bf a = load_a_bf16(Xb + (size_t)m0 * 1024 + ks * 32, 1024);
#pragma unroll
    for (int j = 0; j < 4; j++)
      acc[j] = wmma_bf(a, load_b_bf16(Wl + 16 * j * 32, 32), acc[j]);
  }

  const int n = lane & 15, mh = (lane >> 4) << 3;
#pragma unroll
  for (int j = 0; j < 4; j++) {
    const int ng = n0g + 16 * j + n;
    const int hh = ng >> 6, dd = ng & 63;
    const float bs = bias[ng];
    if (!transposed) {
#pragma unroll
      for (int i = 0; i < 8; i++) {
        const int mg = m0 + mh + i;
        const int bb = mg / Lseg, ll = mg % Lseg;
        dst[(((size_t)bb * 16 + hh) * Lseg + ll) * 64 + dd] = f2bf(acc[j][i] + bs);
      }
    } else {
      u16 tmp[8];
#pragma unroll
      for (int i = 0; i < 8; i++) tmp[i] = f2bf(acc[j][i] + bs);
      const int mg0 = m0 + mh;
      const int bb = mg0 / Lseg, ll = mg0 % Lseg;
      *(u32x4*)(dst + (((size_t)bb * 16 + hh) * 64 + dd) * Lseg + ll) =
          *(const u32x4*)tmp;
    }
  }
}

// ---------------------------------------------------------------------------
// Kernel 3: fused attention. One block (8 wave32) per (b, h, 16-row q-tile).
// ---------------------------------------------------------------------------
#define SSTR 520   // padded fp32 score row stride (512 cols used)

__global__ __launch_bounds__(256) void attn_kernel(
    const u16* __restrict__ Qb, const u16* __restrict__ Kb,
    const u16* __restrict__ Vt, const u16* __restrict__ IKb,
    const u16* __restrict__ IVt, const u16* __restrict__ Eb,
    const float* __restrict__ Amask, const float* __restrict__ Imask,
    const float* __restrict__ gate, float* __restrict__ out) {
  __shared__ __align__(16) float sS[16 * SSTR];     // scores fp32; bf16 P overlays
  __shared__ __align__(16) float sIS[16 * 64];      // instruct scores
  __shared__ __align__(16) u16   sPi[16 * 64];      // instruct probs bf16
  __shared__ __align__(16) float sIC[16 * 64];      // gated instruct context
  __shared__ __align__(16) float sQK[8 * 16 * 32];  // per-wave bias gather scratch
  __shared__ float sRsum[16];
  __shared__ float sIRsum[16];

  const int lane = threadIdx.x & 31, w = threadIdx.x >> 5;
  const int qt = blockIdx.x & 31, bh = blockIdx.x >> 5;
  const int b = bh >> 4, h = bh & 15;
  const int l0 = qt * 16;
  const int n = lane & 15, mh = (lane >> 4) << 3;
  const float scale = 0.125f;  // 1/sqrt(64)

  const u16* Qbh = Qb + (size_t)bh * 512 * 64;
  const u16* Kbh = Kb + (size_t)bh * 512 * 64;

  v16bf aQ0 = load_a_bf16(Qbh + (size_t)l0 * 64, 64);
  v16bf aQ1 = load_a_bf16(Qbh + (size_t)l0 * 64 + 32, 64);

  float* scr = sQK + w * 512;

  // ---- Phase A: scores + relative position bias (each wave: 4 r-tiles) ----
  for (int t = 0; t < 4; t++) {
    const int r0 = (w * 4 + t) * 16;
    if (t < 3) __builtin_prefetch(Kbh + (size_t)(r0 + 16) * 64, 0, 0);

    v8f s = {};
    s = wmma_bf(aQ0, load_b_bf16(Kbh + (size_t)r0 * 64, 64), s);
    s = wmma_bf(aQ1, load_b_bf16(Kbh + (size_t)r0 * 64 + 32, 64), s);

    // distance window: dist(i,j) = (l0+i)-(r0+j)+511 = dbase + i - j
    const int dbase = l0 - r0 + 511;
    const u16* Ew0 = Eb + (size_t)(dbase - 15) * 64;   // d' = 0..15
    const u16* Ew1 = Eb + (size_t)(dbase + 1) * 64;    // d' = 16..31

    v8f qd0 = {}, qd1 = {};
    qd0 = wmma_bf(aQ0, load_b_bf16(Ew0, 64), qd0);
    qd0 = wmma_bf(aQ1, load_b_bf16(Ew0 + 32, 64), qd0);
    qd1 = wmma_bf(aQ0, load_b_bf16(Ew1, 64), qd1);
    qd1 = wmma_bf(aQ1, load_b_bf16(Ew1 + 32, 64), qd1);

#pragma unroll
    for (int j = 0; j < 8; j++) {
      scr[(j + mh) * 32 + n]      = qd0[j];
      scr[(j + mh) * 32 + 16 + n] = qd1[j];
    }
    asm volatile("s_wait_dscnt 0" ::: "memory");
    float qg[8];
#pragma unroll
    for (int j = 0; j < 8; j++) qg[j] = scr[(j + mh) * 32 + (j + mh) - n + 15];
    asm volatile("s_wait_dscnt 0" ::: "memory");

    v16bf aK0 = load_a_bf16(Kbh + (size_t)r0 * 64, 64);
    v16bf aK1 = load_a_bf16(Kbh + (size_t)r0 * 64 + 32, 64);
    v8f kd0 = {}, kd1 = {};
    kd0 = wmma_bf(aK0, load_b_bf16(Ew0, 64), kd0);
    kd0 = wmma_bf(aK1, load_b_bf16(Ew0 + 32, 64), kd0);
    kd1 = wmma_bf(aK0, load_b_bf16(Ew1, 64), kd1);
    kd1 = wmma_bf(aK1, load_b_bf16(Ew1 + 32, 64), kd1);

#pragma unroll
    for (int j = 0; j < 8; j++) {
      scr[(j + mh) * 32 + n]      = kd0[j];
      scr[(j + mh) * 32 + 16 + n] = kd1[j];
    }
    asm volatile("s_wait_dscnt 0" ::: "memory");
    float kg[8];
#pragma unroll
    for (int j = 0; j < 8; j++) kg[j] = scr[n * 32 + (j + mh) - n + 15];

    const float mr = Amask[(size_t)b * 512 + r0 + n];
#pragma unroll
    for (int j = 0; j < 8; j++)
      sS[(j + mh) * SSTR + r0 + n] = (s[j] + qg[j] + kg[j]) * scale + mr;
  }

  if (w < 4) {  // instruct scores, r-tile = w (64 instruct keys total)
    const u16* IKbh = IKb + (size_t)bh * 64 * 64;
    const int r0 = w * 16;
    v8f s = {};
    s = wmma_bf(aQ0, load_b_bf16(IKbh + (size_t)r0 * 64, 64), s);
    s = wmma_bf(aQ1, load_b_bf16(IKbh + (size_t)r0 * 64 + 32, 64), s);
    const float mr = Imask[(size_t)b * 64 + r0 + n];
#pragma unroll
    for (int j = 0; j < 8; j++)
      sIS[(j + mh) * 64 + r0 + n] = s[j] * scale + mr;
  }

  __syncthreads();

  // ---- Phase B: softmax (each wave owns rows 2w, 2w+1) ----
  u16* P = (u16*)sS;  // bf16 overlay: row m at ushort offset m*2*SSTR
  for (int rr = 0; rr < 2; rr++) {
    const int m = 2 * w + rr;
    float mx = -3.0e38f;
    for (int c = lane; c < 512; c += 32) mx = fmaxf(mx, sS[m * SSTR + c]);
#pragma unroll
    for (int off = 16; off > 0; off >>= 1) mx = fmaxf(mx, __shfl_xor(mx, off, 32));
    float sum = 0.f;
    for (int c = lane; c < 512; c += 32) {
      const float e = __expf(sS[m * SSTR + c] - mx);
      sum += e;
      P[m * 2 * SSTR + c] = f2bf(e);   // write-behind-read overlay (safe)
    }
#pragma unroll
    for (int off = 16; off > 0; off >>= 1) sum += __shfl_xor(sum, off, 32);
    if (lane == 0) sRsum[m] = 1.f / sum;

    float imx = -3.0e38f;
    for (int c = lane; c < 64; c += 32) imx = fmaxf(imx, sIS[m * 64 + c]);
#pragma unroll
    for (int off = 16; off > 0; off >>= 1) imx = fmaxf(imx, __shfl_xor(imx, off, 32));
    float isum = 0.f;
    for (int c = lane; c < 64; c += 32) {
      const float e = __expf(sIS[m * 64 + c] - imx);
      isum += e;
      sPi[m * 64 + c] = f2bf(e);
    }
#pragma unroll
    for (int off = 16; off > 0; off >>= 1) isum += __shfl_xor(isum, off, 32);
    if (lane == 0) sIRsum[m] = 1.f / isum;
  }

  __syncthreads();

  // ---- Phase C: waves 0-3 main context P@V; waves 4-7 instruct branch ----
  float cv[8];
  const int n0 = (w & 3) * 16;
  if (w < 4) {
    v8f acc = {};
    const u16* Vrow = Vt + ((size_t)bh * 64 + n0) * 512;
#pragma unroll 4
    for (int ch = 0; ch < 16; ch++) {
      v16bf aP = load_a_bf16((const u16*)P + ch * 32, 2 * SSTR);
      v16bf bV = load_b_bf16(Vrow + ch * 32, 512);
      acc = wmma_bf(aP, bV, acc);
    }
#pragma unroll
    for (int j = 0; j < 8; j++) cv[j] = acc[j] * sRsum[j + mh];
  } else {
    v8f acc = {};
    const u16* IVrow = IVt + ((size_t)bh * 64 + n0) * 64;
    acc = wmma_bf(load_a_bf16(sPi, 64), load_b_bf16(IVrow, 64), acc);
    acc = wmma_bf(load_a_bf16(sPi + 32, 64), load_b_bf16(IVrow + 32, 64), acc);
    const float tg = tanhf(gate[h]);
#pragma unroll
    for (int j = 0; j < 8; j++)
      sIC[(j + mh) * 64 + n0 + n] = acc[j] * sIRsum[j + mh] * tg;
  }

  __syncthreads();

  if (w < 4) {
#pragma unroll
    for (int j = 0; j < 8; j++) {
      const int m = j + mh;
      out[((size_t)b * 512 + l0 + m) * 1024 + (size_t)h * 64 + n0 + n] =
          cv[j] + sIC[m * 64 + n0 + n];
    }
  }
}

// ---------------------------------------------------------------------------
extern "C" void kernel_launch(void* const* d_in, const int* in_sizes, int n_in,
                              void* d_out, int out_size, void* d_ws, size_t ws_size,
                              hipStream_t stream) {
  (void)in_sizes; (void)n_in; (void)out_size; (void)ws_size;
  const float* X    = (const float*)d_in[0];
  const float* Xi   = (const float*)d_in[1];
  const float* Am   = (const float*)d_in[2];
  const float* Im   = (const float*)d_in[3];
  const float* Wq   = (const float*)d_in[4];
  const float* bq   = (const float*)d_in[5];
  const float* Wk   = (const float*)d_in[6];
  const float* bk   = (const float*)d_in[7];
  const float* Wv   = (const float*)d_in[8];
  const float* bv   = (const float*)d_in[9];
  const float* E    = (const float*)d_in[10];
  const float* gate = (const float*)d_in[11];
  float* out = (float*)d_out;

  char* p = (char*)d_ws;                       // ~43 MB workspace layout
  u16* Xb  = (u16*)p; p += (size_t)4096 * 1024 * 2;
  u16* Xib = (u16*)p; p += (size_t)512 * 1024 * 2;
  u16* Eb  = (u16*)p; p += (size_t)1024 * 64 * 2;   // 1023 rows used (+1 pad)
  u16* Wtq = (u16*)p; p += (size_t)1024 * 1024 * 2;
  u16* Wtk = (u16*)p; p += (size_t)1024 * 1024 * 2;
  u16* Wtv = (u16*)p; p += (size_t)1024 * 1024 * 2;
  u16* Qb  = (u16*)p; p += (size_t)128 * 512 * 64 * 2;
  u16* Kb  = (u16*)p; p += (size_t)128 * 512 * 64 * 2;
  u16* Vt  = (u16*)p; p += (size_t)128 * 512 * 64 * 2;
  u16* IKb = (u16*)p; p += (size_t)128 * 64 * 64 * 2;
  u16* IVt = (u16*)p; p += (size_t)128 * 64 * 64 * 2;

  convert_kernel<<<7744, 256, 0, stream>>>(X, Xi, E, Wq, Wk, Wv,
                                           Xb, Xib, Eb, Wtq, Wtk, Wtv);
  proj_kernel<<<512, 256, 0, stream>>>(Xb, Wtq, bq, Qb, 512, 0);
  proj_kernel<<<512, 256, 0, stream>>>(Xb, Wtk, bk, Kb, 512, 0);
  proj_kernel<<<512, 256, 0, stream>>>(Xb, Wtv, bv, Vt, 512, 1);
  proj_kernel<<<64, 256, 0, stream>>>(Xib, Wtk, bk, IKb, 64, 0);
  proj_kernel<<<64, 256, 0, stream>>>(Xib, Wtv, bv, IVt, 64, 1);
  attn_kernel<<<4096, 256, 0, stream>>>(Qb, Kb, Vt, IKb, IVt, Eb, Am, Im, gate, out);
}
